// QLoraLinear_82695300317618
// MI455X (gfx1250) — matrix operand order repaired and checked by reading, not compiled, not run on hardware
//
#include <hip/hip_runtime.h>

// Problem constants (match reference)
#define IN_DIM   4096
#define OUT_DIM  4096
#define TOK_DIM  8192
#define QBLOCK   64
#define LORA_SCALING 8.0f   // ALPHA/RANK = 32/4

// Tiling
#define BM 128
#define BN 128
#define BK 32
#define LDA 40   // halfs; 80 bytes row stride (multiple of 16B for b128 LDS ops)
#define LDB 40

typedef __attribute__((ext_vector_type(16))) _Float16 v16h;
typedef __attribute__((ext_vector_type(8)))  _Float16 h8;
typedef __attribute__((ext_vector_type(4)))  _Float16 h4;
typedef __attribute__((ext_vector_type(8)))  float    v8f;

union Frag16 { v16h v; h8 h[2]; };

__constant__ float NF4_TBL[16] = {
    -1.0f, -0.6961928009986877f, -0.5250730514526367f, -0.39491748809814453f,
    -0.28444138169288635f, -0.18477343022823334f, -0.09105003625154495f, 0.0f,
    0.07958029955625534f, 0.16093020141124725f, 0.24611230194568634f,
    0.33791524171829224f, 0.44070982933044434f, 0.5626170039176941f,
    0.7229568362236023f, 1.0f};

// ---------------------------------------------------------------------------
// Kernel 1: t[tok][4] = x[tok,:] @ lora_A[4,:]^T   (memory-bound, rank 4)
// One wave32 per token row; float4 streaming loads; shfl tree reduction.
// ---------------------------------------------------------------------------
__global__ __launch_bounds__(256) void xa_kernel(const float* __restrict__ x,
                                                 const float* __restrict__ lora_A,
                                                 float* __restrict__ t) {
  const int wave = threadIdx.x >> 5;
  const int lane = threadIdx.x & 31;
  const int row  = blockIdx.x * 8 + wave;

  const float4* x4 = (const float4*)(x + (size_t)row * IN_DIM);
  const float4* a4 = (const float4*)lora_A;   // [4][IN_DIM/4]

  float acc0 = 0.f, acc1 = 0.f, acc2 = 0.f, acc3 = 0.f;
  for (int i = lane; i < IN_DIM / 4; i += 32) {
    float4 xv = x4[i];
    float4 a0 = a4[0 * (IN_DIM / 4) + i];
    float4 a1 = a4[1 * (IN_DIM / 4) + i];
    float4 a2 = a4[2 * (IN_DIM / 4) + i];
    float4 a3 = a4[3 * (IN_DIM / 4) + i];
    acc0 += xv.x * a0.x + xv.y * a0.y + xv.z * a0.z + xv.w * a0.w;
    acc1 += xv.x * a1.x + xv.y * a1.y + xv.z * a1.z + xv.w * a1.w;
    acc2 += xv.x * a2.x + xv.y * a2.y + xv.z * a2.z + xv.w * a2.w;
    acc3 += xv.x * a3.x + xv.y * a3.y + xv.z * a3.z + xv.w * a3.w;
  }
  for (int off = 16; off > 0; off >>= 1) {
    acc0 += __shfl_down(acc0, off, 32);
    acc1 += __shfl_down(acc1, off, 32);
    acc2 += __shfl_down(acc2, off, 32);
    acc3 += __shfl_down(acc3, off, 32);
  }
  if (lane == 0) {
    ((float4*)t)[row] = make_float4(acc0, acc1, acc2, acc3);
  }
}

// ---------------------------------------------------------------------------
// Kernel 2: fused NF4-dequant GEMM + bias + LoRA epilogue.
// 256 threads = 8 waves (wave32). Block tile 128x128, K-step 32.
// Wave grid 4(M) x 2(N): each wave owns 32x64 = 2x4 fragments of 16x16.
// LDS ping-pong double-buffer, K-loop unrolled x2 so buffer indices are
// compile-time constants (static LDS offsets, no per-iter address MADs).
// One barrier per K-step; next-tile global loads overlap the WMMAs.
// ---------------------------------------------------------------------------
__global__ __launch_bounds__(256) void qlora_gemm(
    const float* __restrict__ x, const int* __restrict__ q,
    const float* __restrict__ absmax, const float* __restrict__ bias,
    const float* __restrict__ loraB, const float* __restrict__ t,
    float* __restrict__ out) {
  __shared__ _Float16 As[2][BM * LDA];   // x tile,   [m][k] f16
  __shared__ _Float16 Bs[2][BN * LDB];   // W^T tile, [n][k] f16 (dequantized)
  __shared__ float    nf4s[16];

  const int tid = threadIdx.x;
  if (tid < 16) nf4s[tid] = NF4_TBL[tid];

  const int tok0 = blockIdx.y * BM;
  const int n0   = blockIdx.x * BN;

  const int lane  = tid & 31;
  const int wave  = tid >> 5;
  const int waveM = wave & 3;   // * 32 rows
  const int waveN = wave >> 2;  // * 64 cols
  const int mlo   = lane & 15;
  const int hi    = lane >> 4;  // lane half-select

  // Staging assignment: each thread owns half a tile row (16 elements).
  const int srow = tid >> 1;            // 0..127
  const int skk  = (tid & 1) * 16;      // 0 or 16

  const float* xrow  = x + (size_t)(tok0 + srow) * IN_DIM + skk;
  const int*   qrow  = q + (size_t)(n0 + srow) * IN_DIM + skk;
  const float* amrow = absmax + (size_t)(n0 + srow) * (IN_DIM / QBLOCK);

  v8f acc[2][4] = {};

  // Convert + store one staged tile into compile-time buffer BUF.
  auto stage_store = [&](auto bufc, float4 a0, float4 a1, float4 a2, float4 a3,
                         int4 c0, int4 c1, int4 c2, int4 c3, float am) {
    constexpr int BUF = decltype(bufc)::value;
    h4* dA = (h4*)&As[BUF][srow * LDA + skk];
    h4 x0 = {(_Float16)a0.x, (_Float16)a0.y, (_Float16)a0.z, (_Float16)a0.w};
    h4 x1 = {(_Float16)a1.x, (_Float16)a1.y, (_Float16)a1.z, (_Float16)a1.w};
    h4 x2 = {(_Float16)a2.x, (_Float16)a2.y, (_Float16)a2.z, (_Float16)a2.w};
    h4 x3 = {(_Float16)a3.x, (_Float16)a3.y, (_Float16)a3.z, (_Float16)a3.w};
    dA[0] = x0; dA[1] = x1; dA[2] = x2; dA[3] = x3;

    h4* dB = (h4*)&Bs[BUF][srow * LDB + skk];
    h4 w0 = {(_Float16)(nf4s[c0.x & 15] * am), (_Float16)(nf4s[c0.y & 15] * am),
             (_Float16)(nf4s[c0.z & 15] * am), (_Float16)(nf4s[c0.w & 15] * am)};
    h4 w1 = {(_Float16)(nf4s[c1.x & 15] * am), (_Float16)(nf4s[c1.y & 15] * am),
             (_Float16)(nf4s[c1.z & 15] * am), (_Float16)(nf4s[c1.w & 15] * am)};
    h4 w2 = {(_Float16)(nf4s[c2.x & 15] * am), (_Float16)(nf4s[c2.y & 15] * am),
             (_Float16)(nf4s[c2.z & 15] * am), (_Float16)(nf4s[c2.w & 15] * am)};
    h4 w3 = {(_Float16)(nf4s[c3.x & 15] * am), (_Float16)(nf4s[c3.y & 15] * am),
             (_Float16)(nf4s[c3.z & 15] * am), (_Float16)(nf4s[c3.w & 15] * am)};
    dB[0] = w0; dB[1] = w1; dB[2] = w2; dB[3] = w3;
  };

  // Fragment loads + 8 WMMAs on compile-time buffer BUF.
  auto compute_tile = [&](auto bufc) {
    constexpr int BUF = decltype(bufc)::value;
    Frag16 a[2], b[4];
#pragma unroll
    for (int fm = 0; fm < 2; ++fm) {
      const _Float16* p = &As[BUF][(waveM * 32 + fm * 16 + mlo) * LDA + hi * 8];
      a[fm].h[0] = *(const h8*)(p);        // K = hi*8 .. +8
      a[fm].h[1] = *(const h8*)(p + 16);   // K = 16+hi*8 .. +8
    }
#pragma unroll
    for (int fn = 0; fn < 4; ++fn) {
      const _Float16* p = &Bs[BUF][(waveN * 64 + fn * 16 + mlo) * LDB + hi * 8];
      b[fn].h[0] = *(const h8*)(p);        // K = hi*8 .. +8
      b[fn].h[1] = *(const h8*)(p + 16);   // K = 16+hi*8 .. +8
    }
#pragma unroll
    for (int fm = 0; fm < 2; ++fm)
#pragma unroll
      for (int fn = 0; fn < 4; ++fn)
        acc[fm][fn] = __builtin_amdgcn_wmma_f32_16x16x32_f16(
            false, a[fm].v, false, b[fn].v, (short)0, acc[fm][fn], false, false);
  };

  using B0 = std::integral_constant<int, 0>;
  using B1 = std::integral_constant<int, 1>;

  // ---- prologue: fetch + stage tile 0 into buf0 ----
  {
    const float4* xs = (const float4*)(xrow);
    float4 xv0 = xs[0], xv1 = xs[1], xv2 = xs[2], xv3 = xs[3];
    const int4* qs = (const int4*)(qrow);
    int4 qv0 = qs[0], qv1 = qs[1], qv2 = qs[2], qv3 = qs[3];
    float amax = amrow[0];
    __syncthreads();  // nf4s visible before first dequant
    stage_store(B0{}, xv0, xv1, xv2, xv3, qv0, qv1, qv2, qv3, amax);
  }
  __syncthreads();

  for (int k0 = 0; k0 < IN_DIM; k0 += 2 * BK) {
    // ---- phase A: compute buf0 (tile k0); stage tile k0+BK -> buf1 ----
    {
      const float4* xs = (const float4*)(xrow + k0 + BK);
      float4 nx0 = xs[0], nx1 = xs[1], nx2 = xs[2], nx3 = xs[3];
      const int4* qs = (const int4*)(qrow + k0 + BK);
      int4 nq0 = qs[0], nq1 = qs[1], nq2 = qs[2], nq3 = qs[3];
      float nam = amrow[(k0 + BK) >> 6];

      compute_tile(B0{});
      stage_store(B1{}, nx0, nx1, nx2, nx3, nq0, nq1, nq2, nq3, nam);
      __syncthreads();
    }
    // ---- phase B: compute buf1 (tile k0+BK); stage tile k0+2BK -> buf0 ----
    {
      const bool has_next = (k0 + 2 * BK) < IN_DIM;
      float4 nx0, nx1, nx2, nx3;
      int4   nq0, nq1, nq2, nq3;
      float  nam;
      if (has_next) {
        const float4* xs = (const float4*)(xrow + k0 + 2 * BK);
        nx0 = xs[0]; nx1 = xs[1]; nx2 = xs[2]; nx3 = xs[3];
        const int4* qs = (const int4*)(qrow + k0 + 2 * BK);
        nq0 = qs[0]; nq1 = qs[1]; nq2 = qs[2]; nq3 = qs[3];
        nam = amrow[(k0 + 2 * BK) >> 6];
      }
      compute_tile(B1{});
      if (has_next) {
        stage_store(B0{}, nx0, nx1, nx2, nx3, nq0, nq1, nq2, nq3, nam);
      }
      __syncthreads();
    }
  }

  // ---- epilogue: + bias + LORA_SCALING * (t @ loraB^T) ----
  const float4* t4 = (const float4*)t;       // [TOK][4]
  const float4* b4 = (const float4*)loraB;   // [OUT][4]
  float4 bb[4];
  float  bi[4];
  int    coln[4];
#pragma unroll
  for (int fn = 0; fn < 4; ++fn) {
    coln[fn] = n0 + waveN * 64 + fn * 16 + mlo;
    bb[fn]   = b4[coln[fn]];
    bi[fn]   = bias[coln[fn]];
  }
#pragma unroll
  for (int fm = 0; fm < 2; ++fm) {
#pragma unroll
    for (int vr = 0; vr < 8; ++vr) {
      const int row = tok0 + waveM * 32 + fm * 16 + hi * 8 + vr;  // M = vr + 8*hi
      const float4 tr = t4[row];
      float* orow = out + (size_t)row * OUT_DIM;
#pragma unroll
      for (int fn = 0; fn < 4; ++fn) {
        const float d = tr.x * bb[fn].x + tr.y * bb[fn].y +
                        tr.z * bb[fn].z + tr.w * bb[fn].w;
        orow[coln[fn]] = acc[fm][fn][vr] + bi[fn] + LORA_SCALING * d;
      }
    }
  }
}

// ---------------------------------------------------------------------------
extern "C" void kernel_launch(void* const* d_in, const int* in_sizes, int n_in,
                              void* d_out, int out_size, void* d_ws, size_t ws_size,
                              hipStream_t stream) {
  const float* x      = (const float*)d_in[0];
  const int*   q      = (const int*)  d_in[1];
  const float* absmax = (const float*)d_in[2];
  const float* bias   = (const float*)d_in[3];
  const float* lora_A = (const float*)d_in[4];
  const float* lora_B = (const float*)d_in[5];
  float* out = (float*)d_out;
  float* t   = (float*)d_ws;  // TOK_DIM * 4 floats = 128 KB scratch

  xa_kernel<<<dim3(TOK_DIM / 8), 256, 0, stream>>>(x, lora_A, t);
  qlora_gemm<<<dim3(OUT_DIM / BN, TOK_DIM / BM), 256, 0, stream>>>(
      x, q, absmax, bias, lora_B, t, out);
}
